// MultiHeadAttentionWithSWA_28286654611473
// MI455X (gfx1250) — matrix operand order repaired and checked by reading, not compiled, not run on hardware
//
#include <hip/hip_runtime.h>
#include <hip/hip_bf16.h>

// ---------------------------------------------------------------------------
// MI455X (gfx1250) multi-head attention with sliding window, bf16 WMMA path,
// with Tensor Data Mover (TDM) tile staging where available.
// b=2, s=2048, d=2048, H=16, Dh=128, W=2048 (window == seq -> causal).
// ---------------------------------------------------------------------------

typedef __attribute__((ext_vector_type(16))) __bf16 v16bf;
typedef __attribute__((ext_vector_type(8)))  float  v8f;
typedef __attribute__((ext_vector_type(4)))  unsigned int u32x4;
typedef __attribute__((ext_vector_type(8)))  int i32x8;
typedef __attribute__((ext_vector_type(4)))  int i32x4;

union Frag16 { v16bf v; unsigned short u[16]; };

#define SEQ    2048
#define DMODEL 2048
#define NHEAD  16
#define DHEAD  128
#define SWIN   2048

#if defined(__has_builtin)
#if __has_builtin(__builtin_amdgcn_tensor_load_to_lds) && \
    __has_builtin(__builtin_amdgcn_s_wait_tensorcnt)
#define HAVE_TDM 1
#endif
#endif
#ifndef HAVE_TDM
#define HAVE_TDM 0
#endif

static __device__ __forceinline__ unsigned short f32_bf16(float f) {
    unsigned int u = __float_as_uint(f);
    u += 0x7FFFu + ((u >> 16) & 1u);          // round-to-nearest-even
    return (unsigned short)(u >> 16);
}

// A-fragment (16x32 bf16, MxK) K index for element e, lane-half `half`:
// element e holds K = e + 16*half (contiguous, matching B fragments).
static __device__ __forceinline__ int a_kidx(int e, int half) {
    return e + (half << 4);
}

#if HAVE_TDM
// Issue a TDM 2D tile load: bf16 tile (tile_x contiguous elems, tile_y rows)
// from global row-major tensor with row stride `stride_elems` into LDS.
// D# layout per CDNA5 ISA ch.8 (groups 2/3 zero -> 2D tensor).
static __device__ __forceinline__ void tdm_load_2d(unsigned lds_off,
                                                   const void* gptr,
                                                   unsigned tile_x,
                                                   unsigned tile_y,
                                                   unsigned stride_elems) {
    unsigned long long ga = (unsigned long long)(size_t)gptr;
    u32x4 g0;
    g0.x = 1u;                                           // count=1, user mode
    g0.y = lds_off;                                      // lds_addr (bytes)
    g0.z = (unsigned)ga;                                 // global_addr[31:0]
    g0.w = (unsigned)((ga >> 32) & 0x01FFFFFFull)        // global_addr[56:32]
         | (2u << 30);                                   // type = 2 ("image")
    i32x8 g1;
    g1[0] = 0x00010000;                                  // data_size = 2 bytes
    g1[1] = (int)((tile_x & 0xFFFFu) << 16);             // tensor_dim0[15:0]
    g1[2] = (int)((tile_x >> 16) | ((tile_y & 0xFFFFu) << 16)); // d0hi|d1lo
    g1[3] = (int)((tile_y >> 16) | (tile_x << 16));      // d1hi | tile_dim0
    g1[4] = (int)tile_y;                                 // tile_dim1 (dim2=0)
    g1[5] = (int)stride_elems;                           // tensor_dim0_stride
    g1[6] = 0;
    g1[7] = 0;
    i32x4 z4 = {0, 0, 0, 0};
#if __clang_major__ >= 23
    i32x8 z8 = {0, 0, 0, 0, 0, 0, 0, 0};
    __builtin_amdgcn_tensor_load_to_lds(g0, g1, z4, z4, z8, 0);
#else
    __builtin_amdgcn_tensor_load_to_lds(g0, g1, z4, z4, 0);
#endif
}
#endif

// ---------------------------------------------------------------------------
// fp32 -> bf16 conversion (4 elements / thread)
// ---------------------------------------------------------------------------
__global__ void cvt_f32_bf16(const float* __restrict__ in,
                             unsigned short* __restrict__ out, int n) {
    int i = (blockIdx.x * blockDim.x + threadIdx.x) * 4;
    if (i + 3 < n) {
        float4 f = *(const float4*)(in + i);
        ushort4 o;
        o.x = f32_bf16(f.x); o.y = f32_bf16(f.y);
        o.z = f32_bf16(f.z); o.w = f32_bf16(f.w);
        *(ushort4*)(out + i) = o;
    }
}

// ---------------------------------------------------------------------------
// bf16 GEMM, C = A(MxK) * B(KxN). 128x128 block tile, 256 threads (8 waves).
// Wave layout 2x4: each wave computes 64x32 of C = 4x2 WMMA 16x16 fragments.
// A tile staged by TDM when available; B tile transposed manually into LDS.
// ---------------------------------------------------------------------------
__global__ __launch_bounds__(256)
void gemm_bf16_128x128(const unsigned short* __restrict__ A,
                       const unsigned short* __restrict__ Bm,
                       unsigned short* __restrict__ Cb,
                       float* __restrict__ Cf,
                       int M, int N, int K, int store_bf16) {
    __shared__ __align__(16) unsigned short As[128][32];   // [m][k]
    __shared__ __align__(16) unsigned short Bs[128][32];   // [n][k] (transposed)

    const int t    = threadIdx.x;
    const int lane = t & 31;
    const int w    = t >> 5;
    const int half = lane >> 4;
    const int l16  = lane & 15;
    const int bm   = blockIdx.y * 128;
    const int bn   = blockIdx.x * 128;
    const int wm   = (w >> 2) * 64;
    const int wn   = (w & 3) * 32;

    v8f zero = {};
    v8f c[4][2];
    for (int mi = 0; mi < 4; mi++)
        for (int ni = 0; ni < 2; ni++) c[mi][ni] = zero;

    for (int k0 = 0; k0 < K; k0 += 32) {
        __syncthreads();
#if HAVE_TDM
        if (w == 0)   // DMA A tile 128x32 bf16 global->LDS (TENSORcnt)
            tdm_load_2d((unsigned)(size_t)&As[0][0],
                        &A[(size_t)bm * K + k0], 32u, 128u, (unsigned)K);
#else
        for (int i = 0; i < 2; i++) {
            int u   = t + i * 256;
            int row = u >> 2;
            int col = (u & 3) << 3;
            *(uint4*)(&As[row][col]) =
                *(const uint4*)(&A[(size_t)(bm + row) * K + k0 + col]);
        }
#endif
        // B tile 32x128, transposed into Bs[n][k]
        for (int i = 0; i < 16; i++) {
            int e = i * 256 + t;
            int k = e >> 7;
            int n = e & 127;
            Bs[n][k] = Bm[(size_t)(k0 + k) * N + bn + n];
        }
        if (k0 + 32 < K)      // gfx1250 global_prefetch_b8 on next B tile
            __builtin_prefetch(&Bm[(size_t)(k0 + 32 + (t >> 3)) * N + bn], 0, 0);
#if HAVE_TDM
        if (w == 0) __builtin_amdgcn_s_wait_tensorcnt(0);
#endif
        __syncthreads();

        Frag16 bf[2];
        for (int ni = 0; ni < 2; ni++) {
            int n = wn + ni * 16 + l16;
            for (int e = 0; e < 16; e++)
                bf[ni].u[e] = Bs[n][e + half * 16];
        }
        for (int mi = 0; mi < 4; mi++) {
            Frag16 af;
            int m = wm + mi * 16 + l16;
            for (int e = 0; e < 16; e++)
                af.u[e] = As[m][a_kidx(e, half)];
            for (int ni = 0; ni < 2; ni++)
                c[mi][ni] = __builtin_amdgcn_wmma_f32_16x16x32_bf16(
                    false, af.v, false, bf[ni].v, (short)0, c[mi][ni],
                    false, false);
        }
    }

    for (int mi = 0; mi < 4; mi++)
        for (int ni = 0; ni < 2; ni++)
            for (int r = 0; r < 8; r++) {
                int row = bm + wm + mi * 16 + r + half * 8;
                int col = bn + wn + ni * 16 + l16;
                float val = c[mi][ni][r];
                if (store_bf16) Cb[(size_t)row * N + col] = f32_bf16(val);
                else            Cf[(size_t)row * N + col] = val;
            }
}

// ---------------------------------------------------------------------------
// Flash-attention with sliding window. One block = (b, h, 64-query tile),
// 128 threads = 4 waves, one 16-row stripe per wave. 32-key K/V tiles.
// Q/K/V tiles staged by TDM when available.
// ---------------------------------------------------------------------------
__global__ __launch_bounds__(128)
void attn_swa(const unsigned short* __restrict__ Qg,
              const unsigned short* __restrict__ Kg,
              const unsigned short* __restrict__ Vg,
              unsigned short* __restrict__ Ctx) {
    __shared__ __align__(16) unsigned short Qs[64][128];
    __shared__ __align__(16) unsigned short Ks[32][128];
    __shared__ __align__(16) unsigned short Vs[32][128];
    __shared__ __align__(16) unsigned short Ps[4][16][32];  // per-wave P staging

    const int t    = threadIdx.x;
    const int lane = t & 31;
    const int w    = t >> 5;
    const int half = lane >> 4;
    const int l16  = lane & 15;

    const int q0 = blockIdx.x * 64;
    const int h  = blockIdx.y;
    const int b  = blockIdx.z;

    const unsigned short* Qb = Qg + ((size_t)b * SEQ) * DMODEL + (size_t)h * DHEAD;
    const unsigned short* Kb = Kg + ((size_t)b * SEQ) * DMODEL + (size_t)h * DHEAD;
    const unsigned short* Vb = Vg + ((size_t)b * SEQ) * DMODEL + (size_t)h * DHEAD;

#if HAVE_TDM
    if (w == 0)   // DMA Q tile 64x128 bf16 (head slice, row stride DMODEL)
        tdm_load_2d((unsigned)(size_t)&Qs[0][0],
                    &Qb[(size_t)q0 * DMODEL], 128u, 64u, (unsigned)DMODEL);
#else
    for (int i = 0; i < 8; i++) {
        int u   = i * 128 + t;
        int row = u >> 4;
        int col = (u & 15) << 3;
        *(uint4*)(&Qs[row][col]) =
            *(const uint4*)(&Qb[(size_t)(q0 + row) * DMODEL + col]);
    }
#endif

    float m_i[8], l_i[8];
    v8f zero = {};
    v8f acc[8];
    for (int r = 0; r < 8; r++) { m_i[r] = -3.0e38f; l_i[r] = 0.0f; }
    for (int nb = 0; nb < 8; nb++) acc[nb] = zero;

    const float scale = 0.088388347648318447f;  // 1/sqrt(128)
    const int qw = q0 + w * 16;

    int klo = q0 - (SWIN - 1); if (klo < 0) klo = 0;
    const int kt_lo = klo >> 5;
    const int kt_hi = (q0 + 63) >> 5;

    for (int kt = kt_lo; kt <= kt_hi; kt++) {
        const int kbse = kt * 32;
        __syncthreads();
#if HAVE_TDM
        if (w == 0) {
            tdm_load_2d((unsigned)(size_t)&Ks[0][0],
                        &Kb[(size_t)kbse * DMODEL], 128u, 32u, (unsigned)DMODEL);
            tdm_load_2d((unsigned)(size_t)&Vs[0][0],
                        &Vb[(size_t)kbse * DMODEL], 128u, 32u, (unsigned)DMODEL);
            __builtin_amdgcn_s_wait_tensorcnt(0);  // also covers the Q tile
        }
#else
        for (int i = 0; i < 4; i++) {
            int u   = i * 128 + t;
            int row = u >> 4;
            int col = (u & 15) << 3;
            *(uint4*)(&Ks[row][col]) =
                *(const uint4*)(&Kb[(size_t)(kbse + row) * DMODEL + col]);
            *(uint4*)(&Vs[row][col]) =
                *(const uint4*)(&Vb[(size_t)(kbse + row) * DMODEL + col]);
        }
#endif
        __syncthreads();

        // S = Q * K^T for this wave's 16 rows x 32 keys (two 16x16 C tiles)
        v8f s0 = zero, s1 = zero;
        for (int ds = 0; ds < DHEAD; ds += 32) {
            Frag16 af, b0, b1;
            for (int e = 0; e < 16; e++) {
                int kk = a_kidx(e, half);
                af.u[e] = Qs[w * 16 + l16][ds + kk];
                b0.u[e] = Ks[l16][ds + e + half * 16];
                b1.u[e] = Ks[16 + l16][ds + e + half * 16];
            }
            s0 = __builtin_amdgcn_wmma_f32_16x16x32_bf16(
                false, af.v, false, b0.v, (short)0, s0, false, false);
            s1 = __builtin_amdgcn_wmma_f32_16x16x32_bf16(
                false, af.v, false, b1.v, (short)0, s1, false, false);
        }

        // mask + online softmax; fragment row = r + 8*half, col = l16 (+16)
        for (int r = 0; r < 8; r++) {
            int q  = qw + r + half * 8;
            int k0 = kbse + l16;
            int k1 = kbse + 16 + l16;
            float v0 = s0[r] * scale;
            float v1 = s1[r] * scale;
            if (k0 > q || (q - k0) >= SWIN) v0 = -3.0e38f;
            if (k1 > q || (q - k1) >= SWIN) v1 = -3.0e38f;
            float rm = fmaxf(v0, v1);
            for (int off = 8; off >= 1; off >>= 1)
                rm = fmaxf(rm, __shfl_xor(rm, off, 32));
            float mnew = fmaxf(m_i[r], rm);
            float corr = __expf(m_i[r] - mnew);
            float p0 = __expf(v0 - mnew);
            float p1 = __expf(v1 - mnew);
            float rs = p0 + p1;
            for (int off = 8; off >= 1; off >>= 1)
                rs += __shfl_xor(rs, off, 32);
            l_i[r] = l_i[r] * corr + rs;
            m_i[r] = mnew;
            for (int nb = 0; nb < 8; nb++) acc[nb][r] *= corr;
            // stage P (C layout) into per-wave LDS for A-layout reload
            Ps[w][r + half * 8][l16]      = f32_bf16(p0);
            Ps[w][r + half * 8][16 + l16] = f32_bf16(p1);
        }

        // O += P * V   (A = P 16x32, B = V 32x16 per 16-wide dh block)
        Frag16 pf;
        for (int e = 0; e < 16; e++)
            pf.u[e] = Ps[w][l16][a_kidx(e, half)];
        for (int nb = 0; nb < 8; nb++) {
            Frag16 bv;
            for (int e = 0; e < 16; e++)
                bv.u[e] = Vs[e + half * 16][nb * 16 + l16];
            acc[nb] = __builtin_amdgcn_wmma_f32_16x16x32_bf16(
                false, pf.v, false, bv.v, (short)0, acc[nb], false, false);
        }
    }

    // normalize and store ctx (bf16) at [b, q, h*128 + dh]
    unsigned short* Cb = Ctx + ((size_t)b * SEQ) * DMODEL + (size_t)h * DHEAD;
    for (int r = 0; r < 8; r++) {
        float inv = 1.0f / l_i[r];
        int q = qw + r + half * 8;
        for (int nb = 0; nb < 8; nb++)
            Cb[(size_t)q * DMODEL + nb * 16 + l16] = f32_bf16(acc[nb][r] * inv);
    }
}

// ---------------------------------------------------------------------------
// Launch: cvt -> 3x GEMM (QKV) -> attention -> GEMM (Wo)
// ---------------------------------------------------------------------------
extern "C" void kernel_launch(void* const* d_in, const int* in_sizes, int n_in,
                              void* d_out, int out_size, void* d_ws, size_t ws_size,
                              hipStream_t stream) {
    const float* x  = (const float*)d_in[0];
    const float* Wq = (const float*)d_in[1];
    const float* Wk = (const float*)d_in[2];
    const float* Wv = (const float*)d_in[3];
    const float* Wo = (const float*)d_in[4];
    float* out = (float*)d_out;

    const size_t nX = (size_t)2 * SEQ * DMODEL;   // 8388608
    const size_t nW = (size_t)DMODEL * DMODEL;    // 4194304

    unsigned short* ws = (unsigned short*)d_ws;
    size_t off = 0;
    unsigned short* xb  = ws + off; off += nX;
    unsigned short* Wqb = ws + off; off += nW;
    unsigned short* Wkb = ws + off; off += nW;
    unsigned short* Wvb = ws + off; off += nW;
    unsigned short* Wob = ws + off; off += nW;
    unsigned short* Qb  = ws + off; off += nX;
    unsigned short* Kb  = ws + off; off += nX;
    unsigned short* Vb  = ws + off; off += nX;
    unsigned short* Cxb = ws + off; off += nX;   // ~112 MB total scratch

    cvt_f32_bf16<<<dim3((unsigned)(nX / 1024)), 256, 0, stream>>>(x,  xb,  (int)nX);
    cvt_f32_bf16<<<dim3((unsigned)(nW / 1024)), 256, 0, stream>>>(Wq, Wqb, (int)nW);
    cvt_f32_bf16<<<dim3((unsigned)(nW / 1024)), 256, 0, stream>>>(Wk, Wkb, (int)nW);
    cvt_f32_bf16<<<dim3((unsigned)(nW / 1024)), 256, 0, stream>>>(Wv, Wvb, (int)nW);
    cvt_f32_bf16<<<dim3((unsigned)(nW / 1024)), 256, 0, stream>>>(Wo, Wob, (int)nW);

    dim3 gg(DMODEL / 128, (2 * SEQ) / 128);   // 16 x 32
    gemm_bf16_128x128<<<gg, 256, 0, stream>>>(xb, Wqb, Qb, nullptr,
                                              2 * SEQ, DMODEL, DMODEL, 1);
    gemm_bf16_128x128<<<gg, 256, 0, stream>>>(xb, Wkb, Kb, nullptr,
                                              2 * SEQ, DMODEL, DMODEL, 1);
    gemm_bf16_128x128<<<gg, 256, 0, stream>>>(xb, Wvb, Vb, nullptr,
                                              2 * SEQ, DMODEL, DMODEL, 1);

    attn_swa<<<dim3(SEQ / 64, NHEAD, 2), 128, 0, stream>>>(Qb, Kb, Vb, Cxb);

    gemm_bf16_128x128<<<gg, 256, 0, stream>>>(Cxb, Wob, nullptr, out,
                                              2 * SEQ, DMODEL, DMODEL, 0);
}
